// MixerAttention_49520972923332
// MI455X (gfx1250) — matrix operand order repaired and checked by reading, not compiled, not run on hardware
//
#include <hip/hip_runtime.h>
#include <hip/hip_bf16.h>

// ---------------------------------------------------------------------------
// MI455X (gfx1250) GQA prefill:  y = Attn(RoPE(RMS(x@WqT)), RoPE(RMS(x@WkT)), x@WvT)
// B=2 T=2048 D=2048 H=16 HKV=4 DH=128 G=4, causal, fp32 in/out.
// Matrix-core bound (~120 GFLOP vs ~100MB traffic) -> f16 WMMA, f32 accum.
// Pipeline: [cvt x,W -> f16] -> [QKV GEMM, f16 operands] -> [RMS+RoPE pack]
//           -> [flash attention].
// ---------------------------------------------------------------------------

#define B_    2
#define T_    2048
#define D_    2048
#define H_    16
#define HKV_  4
#define DH_   128
#define NQKV_ 3072   // 2048 q + 512 k + 512 v channels

typedef __attribute__((ext_vector_type(16))) _Float16 v16h;
typedef __attribute__((ext_vector_type(8)))  _Float16 v8h;
typedef __attribute__((ext_vector_type(8)))  float    v8f;
typedef __attribute__((ext_vector_type(4)))  float    v4f;

__device__ __forceinline__ v8h load8h(const _Float16* p) {
  return *(const v8h*)p;   // 16B-aligned at every call site
}

__device__ __forceinline__ v16h cat16(v8h a, v8h b) {
  return __builtin_shufflevector(a, b, 0,1,2,3,4,5,6,7,8,9,10,11,12,13,14,15);
}

__device__ __forceinline__ v8f wmma_f16(v16h a, v16h b, v8f c) {
  // D = A(16x32 f16) * B(32x16 f16) + C(16x16 f32)
  return __builtin_amdgcn_wmma_f32_16x16x32_f16(false, a, false, b,
                                                (short)0, c, false, false);
}

// ---------------------------------------------------------------------------
// Kernel 0: bulk fp32 -> f16 conversion (8 elements / thread, b128 in, b128 out)
// ---------------------------------------------------------------------------
__global__ __launch_bounds__(256) void cvt16_kernel(
    const float* __restrict__ src, _Float16* __restrict__ dst, int n8)
{
  int i = blockIdx.x * 256 + threadIdx.x;
  if (i >= n8) return;
  v4f a = ((const v4f*)src)[2 * i];
  v4f b = ((const v4f*)src)[2 * i + 1];
  v8h r = { (_Float16)a.x, (_Float16)a.y, (_Float16)a.z, (_Float16)a.w,
            (_Float16)b.x, (_Float16)b.y, (_Float16)b.z, (_Float16)b.w };
  ((v8h*)dst)[i] = r;
}

// ---------------------------------------------------------------------------
// Kernel 1: fused QKV projection GEMM on f16 operands.
//   raw[m][n] = x16[m][:] . w16[n][:]   (f32 accumulate)
//   M = B*T = 4096, N = 3072 (q|k|v concat in w16), K = D = 2048.
//   Wave tile: 16(M) x 128(N), 8 waves/block -> 16 x 1024.  Grid: (3, 256).
// A layout (16-bit 16x32): lane&15 = M row; lanes<16 hold K 0..15,
// lanes>=16 hold K 16..31.  B layout (32x16): lane&15 = N col;
// lanes<16 hold K 0..15, lanes>=16 hold K 16..31 (contiguous in d).
// Inner loop is pure global_load_b128 -> v_wmma (no cvt).
// ---------------------------------------------------------------------------
__global__ __launch_bounds__(256) void qkv_gemm_kernel(
    const _Float16* __restrict__ x16, const _Float16* __restrict__ w16,
    float* __restrict__ raw)
{
  const int lane  = threadIdx.x & 31;
  const int wv    = threadIdx.x >> 5;
  const int m0    = blockIdx.y * 16;
  const int nbase = blockIdx.x * 1024 + wv * 128;
  const int lm    = lane & 15;
  const int khiA  = (lane < 16) ? 0 : 16;
  const int khiB  = (lane < 16) ? 0 : 16;

  const _Float16* xrow = x16 + (size_t)(m0 + lm) * D_;
  const _Float16* wrow[8];
#pragma unroll
  for (int j = 0; j < 8; ++j)
    wrow[j] = w16 + (size_t)(nbase + 16 * j + lm) * D_;

  v8f acc[8] = {};

  for (int k0 = 0; k0 < D_; k0 += 32) {
    __builtin_prefetch(xrow + k0 + 1024, 0, 1);     // global_prefetch_b8
    v16h a = cat16(load8h(xrow + k0 + khiA), load8h(xrow + k0 + khiA + 8));
#pragma unroll
    for (int j = 0; j < 8; ++j) {
      v16h bb = cat16(load8h(wrow[j] + k0 + khiB),
                      load8h(wrow[j] + k0 + khiB + 8));
      acc[j] = wmma_f16(a, bb, acc[j]);
    }
  }

  // C/D layout: VGPR r, lane -> row m0 + r + 8*(lane>=16), col n (lane&15)
  const int rbase = m0 + 8 * (lane >> 4);
#pragma unroll
  for (int j = 0; j < 8; ++j) {
    int n = nbase + 16 * j + lm;
#pragma unroll
    for (int r = 0; r < 8; ++r)
      raw[(size_t)(rbase + r) * NQKV_ + n] = acc[j][r];
  }
}

// ---------------------------------------------------------------------------
// Kernel 2: per-head RMS norm + RoPE + f16 pack.  One wave per (b,t,head) row
// of 128.  mode 0: q -> q16[B][H][T][DH]
//          mode 1: k -> k16[B][HKV][T][DH]
//          mode 2: v -> vt [B][HKV][DH][T]   (transposed, no norm/rope)
// ---------------------------------------------------------------------------
__global__ __launch_bounds__(256) void normrope_kernel(
    const float* __restrict__ raw, _Float16* __restrict__ dst, int mode)
{
  const int lane = threadIdx.x & 31;
  const int wv   = threadIdx.x >> 5;
  const int nh   = (mode == 0) ? H_ : HKV_;
  const int col0 = (mode == 0) ? 0 : ((mode == 1) ? 2048 : 2560);

  int r   = blockIdx.x * 8 + wv;
  int b   = r / (T_ * nh);
  int rem = r - b * (T_ * nh);
  int t   = rem / nh;
  int h   = rem - t * nh;

  const float* src = raw + (size_t)(b * T_ + t) * NQKV_ + col0 + h * DH_ + 4 * lane;
  v4f xv = *(const v4f*)src;
  float y[4] = { xv.x, xv.y, xv.z, xv.w };

  if (mode < 2) {
    // RMS over 128 elements (4 per lane, 32 lanes)
    float ss = xv.x*xv.x + xv.y*xv.y + xv.z*xv.z + xv.w*xv.w;
#pragma unroll
    for (int m = 16; m >= 1; m >>= 1) ss += __shfl_xor(ss, m);
    float rinv = rsqrtf(ss * (1.0f / 128.0f) + 1.1920928955078125e-07f);
    float xn[4] = { xv.x*rinv, xv.y*rinv, xv.z*rinv, xv.w*rinv };
    // RoPE: dim i pairs with i^64 -> lane^16.  ip = dim & 63.
#pragma unroll
    for (int jj = 0; jj < 4; ++jj) {
      int   ip   = 4 * (lane & 15) + jj;
      float freq = (float)t * expf(-0.14391156831212787f * (float)ip); // ln(1e4)/64
      float c = cosf(freq), s = sinf(freq);
      float part = __shfl_xor(xn[jj], 16);
      y[jj] = (lane < 16) ? (xn[jj]*c + part*s) : (part*(-s) + xn[jj]*c);
    }
  }

  if (mode == 2) {
#pragma unroll
    for (int jj = 0; jj < 4; ++jj) {
      int dh = 4 * lane + jj;
      dst[((size_t)(b * HKV_ + h) * DH_ + dh) * T_ + t] = (_Float16)y[jj];
    }
  } else {
    _Float16* o = dst + ((size_t)(b * nh + h) * T_ + t) * DH_ + 4 * lane;
#pragma unroll
    for (int jj = 0; jj < 4; ++jj) o[jj] = (_Float16)y[jj];
  }
}

// ---------------------------------------------------------------------------
// Kernel 3: causal flash attention, one wave per 16-row q tile of one (b,hq).
// Per 32-key step: 8 WMMA for S = Q.K^T (f32), online softmax with 16-lane
// shfl row reductions, P repacked C-layout -> A-layout through wave-private
// LDS (same-wave DS ops are in-order: no barrier), 8 WMMA for P.V (V stored
// transposed so B-fragments are contiguous).  Grid: 1024 x 128 threads.
// ---------------------------------------------------------------------------
__global__ __launch_bounds__(128) void attn_kernel(
    const _Float16* __restrict__ q16, const _Float16* __restrict__ k16,
    const _Float16* __restrict__ vt,  float* __restrict__ out)
{
  __shared__ _Float16 psh[4 * 16 * 32];

  const int lane = threadIdx.x & 31;
  const int wv   = threadIdx.x >> 5;
  const int wg   = blockIdx.x * 4 + wv;
  const int qt   = wg & 127;
  const int hq   = (wg >> 7) & 15;
  const int b    = wg >> 11;
  const int q0   = qt * 16;
  const int hkv  = hq >> 2;       // GQA: 4 q heads per kv head
  const int lm   = lane & 15;
  const int hi   = lane >> 4;
  const int khiA = hi * 16;
  const int khiB = hi * 16;

  const _Float16* qbase = q16 + ((size_t)(b * H_   + hq ) * T_ + q0) * DH_;
  const _Float16* kbase = k16 + ((size_t)(b * HKV_ + hkv) * T_) * DH_;
  const _Float16* vbase = vt  + ((size_t)(b * HKV_ + hkv) * DH_) * T_;
  _Float16* pb = psh + wv * 512;

  // Q A-fragments for the 4 DH chunks of 32
  v16h qa[4];
  {
    const _Float16* qrow = qbase + lm * DH_;
#pragma unroll
    for (int c = 0; c < 4; ++c)
      qa[c] = cat16(load8h(qrow + 32 * c + khiA),
                    load8h(qrow + 32 * c + khiA + 8));
  }

  v8f   o[8] = {};
  float mrow[8], lrow[8];
#pragma unroll
  for (int j = 0; j < 8; ++j) { mrow[j] = -3.0e38f; lrow[j] = 0.0f; }

  const float scale = 0.08838834764831845f;   // 1/sqrt(128)
  const int   qgb   = q0 + 8 * hi;            // this lane's row base
  const int   kend  = q0 + 16;

  for (int k0 = 0; k0 < kend; k0 += 32) {
    // ---- S = Q.K^T for two 16-key halves ----
    v8f s0 = {}, s1 = {};
    {
      const _Float16* kr0 = kbase + (size_t)(k0 + lm) * DH_;
      const _Float16* kr1 = kbase + (size_t)(k0 + 16 + lm) * DH_;
#pragma unroll
      for (int c = 0; c < 4; ++c) {
        v16h bb0 = cat16(load8h(kr0 + 32 * c + khiB),
                         load8h(kr0 + 32 * c + khiB + 8));
        s0 = wmma_f16(qa[c], bb0, s0);
      }
#pragma unroll
      for (int c = 0; c < 4; ++c) {
        v16h bb1 = cat16(load8h(kr1 + 32 * c + khiB),
                         load8h(kr1 + 32 * c + khiB + 8));
        s1 = wmma_f16(qa[c], bb1, s1);
      }
    }

    // ---- causal mask + online softmax (row = 16-lane half group) ----
    const int key0 = k0 + lm, key1 = k0 + 16 + lm;
#pragma unroll
    for (int j = 0; j < 8; ++j) {
      int   qg = qgb + j;
      float a0 = (key0 <= qg) ? s0[j] * scale : -3.0e38f;
      float a1 = (key1 <= qg) ? s1[j] * scale : -3.0e38f;
      float mx = fmaxf(a0, a1);
      mx = fmaxf(mx, __shfl_xor(mx, 1));
      mx = fmaxf(mx, __shfl_xor(mx, 2));
      mx = fmaxf(mx, __shfl_xor(mx, 4));
      mx = fmaxf(mx, __shfl_xor(mx, 8));
      float mn   = fmaxf(mrow[j], mx);
      float corr = __expf(mrow[j] - mn);
      float p0   = __expf(a0 - mn);
      float p1   = __expf(a1 - mn);
      float ps   = p0 + p1;
      ps += __shfl_xor(ps, 1);
      ps += __shfl_xor(ps, 2);
      ps += __shfl_xor(ps, 4);
      ps += __shfl_xor(ps, 8);
      lrow[j] = lrow[j] * corr + ps;
      mrow[j] = mn;
#pragma unroll
      for (int d = 0; d < 8; ++d) o[d][j] *= corr;
      int m = j + 8 * hi;
      pb[m * 32 + lm]      = (_Float16)p0;   // C-layout -> LDS
      pb[m * 32 + 16 + lm] = (_Float16)p1;
    }

    // ---- reload P in A-layout (wave-local LDS transpose) ----
    v16h pa = cat16(load8h(pb + lm * 32 + khiA),
                    load8h(pb + lm * 32 + khiA + 8));

    // ---- O += P.V  (V transposed: contiguous over keys) ----
#pragma unroll
    for (int d = 0; d < 8; ++d) {
      const _Float16* vrow = vbase + (size_t)(16 * d + lm) * T_ + k0 + khiB;
      v16h bv = cat16(load8h(vrow), load8h(vrow + 8));
      o[d] = wmma_f16(pa, bv, o[d]);
    }
  }

  // ---- normalize and store fp32 ----
#pragma unroll
  for (int j = 0; j < 8; ++j) {
    float inv = 1.0f / lrow[j];
    int   m   = j + 8 * hi;
    float* orow = out + ((size_t)b * T_ + q0 + m) * (H_ * DH_) + hq * DH_ + lm;
#pragma unroll
    for (int d = 0; d < 8; ++d) orow[16 * d] = o[d][j] * inv;
  }
}

// ---------------------------------------------------------------------------
// Host launcher.  Workspace layout (~76 MB), with time-disjoint overlap:
//   raw fp32 [4096][3072]        @ 0          (50,331,648 B)   live k1 -> k2
//   x16 f16  [4096][2048]        @ 50331648   (16,777,216 B)   live k0 -> k1
//   w16 f16  [3072][2048]        @ 67108864   (12,582,912 B)   live k0 -> k1
//   q16 f16  [B][H][T][DH]       @ 50331648   (16,777,216 B)   live k2 -> k3  (reuses x16)
//   k16 f16  [B][HKV][T][DH]     @ 67108864   ( 4,194,304 B)   live k2 -> k3  (reuses w16)
//   vt  f16  [B][HKV][DH][T]     @ 71303168   ( 4,194,304 B)   live k2 -> k3  (reuses w16)
// ---------------------------------------------------------------------------
extern "C" void kernel_launch(void* const* d_in, const int* in_sizes, int n_in,
                              void* d_out, int out_size, void* d_ws, size_t ws_size,
                              hipStream_t stream)
{
  (void)in_sizes; (void)n_in; (void)out_size; (void)ws_size;
  const float* x  = (const float*)d_in[0];
  const float* Wq = (const float*)d_in[1];
  const float* Wk = (const float*)d_in[2];
  const float* Wv = (const float*)d_in[3];
  float* out = (float*)d_out;

  char* ws = (char*)d_ws;
  float*    raw = (float*)ws;
  _Float16* x16 = (_Float16*)(ws + 50331648u);
  _Float16* w16 = (_Float16*)(ws + 67108864u);
  _Float16* q16 = (_Float16*)(ws + 50331648u);   // reuses x16 (dead after GEMM)
  _Float16* k16 = (_Float16*)(ws + 67108864u);   // reuses w16 (dead after GEMM)
  _Float16* vt  = (_Float16*)(ws + 71303168u);

  // 0) pack operands to f16 (x, then Wq|Wk|Wv concatenated row-major)
  cvt16_kernel<<<dim3(4096), 256, 0, stream>>>(x,  x16,               1048576);
  cvt16_kernel<<<dim3(2048), 256, 0, stream>>>(Wq, w16,                524288);
  cvt16_kernel<<<dim3( 512), 256, 0, stream>>>(Wk, w16 + 4194304u,     131072);
  cvt16_kernel<<<dim3( 512), 256, 0, stream>>>(Wv, w16 + 5242880u,     131072);

  // 1) QKV projection GEMM (pure f16-load -> wmma inner loop)
  qkv_gemm_kernel<<<dim3(3, 256), 256, 0, stream>>>(x16, w16, raw);

  // 2) RMS norm + RoPE + pack / transpose
  normrope_kernel<<<dim3(8192), 256, 0, stream>>>(raw, q16, 0);  // q: B*T*H rows
  normrope_kernel<<<dim3(2048), 256, 0, stream>>>(raw, k16, 1);  // k: B*T*HKV rows
  normrope_kernel<<<dim3(2048), 256, 0, stream>>>(raw, vt,  2);  // v: transpose+cvt

  // 3) causal flash attention
  attn_kernel<<<dim3(1024), 128, 0, stream>>>(q16, k16, vt, out);
}
